// NystromAttention_17609365914144
// MI455X (gfx1250) — matrix operand order repaired and checked by reading, not compiled, not run on hardware
//
#include <hip/hip_runtime.h>

// ---------------------------------------------------------------------------
// Nystrom attention for MI455X (gfx1250, wave32, WMMA bf16 16x16x32).
// Heavy GEMMs: v_wmma_f32_16x16x32_bf16 with bf16 operands staged to LDS via
// gfx1250 async global->LDS copies (ASYNCcnt) when the builtin is available.
// Softmax stats & the 64x64 inverse stay f32 VALU.
// ---------------------------------------------------------------------------

#define DEVFN static __device__ __forceinline__

typedef __bf16        v16bf __attribute__((ext_vector_type(16)));
typedef float         v8f   __attribute__((ext_vector_type(8)));
typedef unsigned int  v8u   __attribute__((ext_vector_type(8)));

constexpr int   B_   = 4;
constexpr int   N_   = 4096;
constexpr int   C_   = 768;
constexpr int   H_   = 12;
constexpr int   D_   = 64;
constexpr int   BH_  = 48;           // B_*H_
constexpr float SCALE_   = 0.125f;   // D^-0.5
constexpr float EPSREG_  = 1e-4f;

// ---- gfx1250 async global->LDS copy (falls back to sync copy if absent) ----
#if __has_builtin(__builtin_amdgcn_global_load_async_to_lds_b128)
#define NYS_ASYNC 1
#endif

typedef int nys_v4i __attribute__((vector_size(16)));

DEVFN void copy16_g2l(const __bf16* g, __bf16* l) {
#ifdef NYS_ASYNC
  __builtin_amdgcn_global_load_async_to_lds_b128(
      (__attribute__((address_space(1))) nys_v4i*)g,
      (__attribute__((address_space(3))) nys_v4i*)l, 0, 0);
#else
  *reinterpret_cast<uint4*>(l) = *reinterpret_cast<const uint4*>(g);
#endif
}

DEVFN void wait_async() {
#ifdef NYS_ASYNC
#if __has_builtin(__builtin_amdgcn_s_wait_asynccnt)
  __builtin_amdgcn_s_wait_asynccnt(0);
#else
  asm volatile("s_wait_asynccnt 0x0" ::: "memory");
#endif
#endif
}

DEVFN v8f vzero8() {
  v8f v;
  for (int i = 0; i < 8; ++i) v[i] = 0.f;
  return v;
}

// A operand: 16(M) x 32(K) bf16, src row-major [row][k], ld in elements.
// gfx1250 layout: lanes 0-15 -> M=lane, K 0-7 (v0-3) & 16-23 (v4-7);
//                 lanes 16-31 -> same M, K 8-15 & 24-31.
DEVFN v16bf load_frag_a(const __bf16* src, int ld) {
  const int lane = threadIdx.x & 31;
  const int row  = lane & 15;
  const int o    = (lane >> 4) << 3;                 // 0 or 8
  const __bf16* p = src + row * ld + o;
  uint4 lo = *reinterpret_cast<const uint4*>(p);      // K = o..o+7
  uint4 hi = *reinterpret_cast<const uint4*>(p + 16); // K = o+16..o+23
  v8u u;
  u[0] = lo.x; u[1] = lo.y; u[2] = lo.z; u[3] = lo.w;
  u[4] = hi.x; u[5] = hi.y; u[6] = hi.z; u[7] = hi.w;
  return __builtin_bit_cast(v16bf, u);
}

// B operand: 32(K) x 16(N) bf16 where source is "NT": row-major [col][k].
// gfx1250 layout: lane = col (mod 16); lanes 0-15 hold K 0-7 & 16-23, lanes 16-31 K 8-15 & 24-31.
DEVFN v16bf load_frag_b(const __bf16* src, int ld) {
  const int lane = threadIdx.x & 31;
  const int col  = lane & 15;
  const int kb   = (lane >> 4) << 3;                 // 0 or 8
  const __bf16* p = src + col * ld + kb;
  uint4 lo = *reinterpret_cast<const uint4*>(p);
  uint4 hi = *reinterpret_cast<const uint4*>(p + 16);
  v8u u;
  u[0] = lo.x; u[1] = lo.y; u[2] = lo.z; u[3] = lo.w;
  u[4] = hi.x; u[5] = hi.y; u[6] = hi.z; u[7] = hi.w;
  return __builtin_bit_cast(v16bf, u);
}

DEVFN v8f wmma_bf16(v16bf a, v16bf b, v8f c) {
  return __builtin_amdgcn_wmma_f32_16x16x32_bf16(false, a, false, b,
                                                 (short)0, c, false, false);
}

// C fragment scalar store helper: m = v + 8*(lane>>4), n = lane&15.
DEVFN void store_cfrag_f32(float* dst, int ld, const v8f& c) {
  const int lane = threadIdx.x & 31;
  const int n  = lane & 15;
  const int mb = (lane >> 4) << 3;
  for (int v = 0; v < 8; ++v) dst[(size_t)(mb + v) * ld + n] = c[v];
}

// ---------------------------------------------------------------------------
// K0: f32 -> bf16 elementwise convert (4 per thread). n % 1024 == 0.
// ---------------------------------------------------------------------------
__global__ __launch_bounds__(256)
void nys_cvt_bf16(const float* __restrict__ src, __bf16* __restrict__ dst) {
  const size_t i = ((size_t)blockIdx.x * 256 + threadIdx.x) * 4;
  const float4 v = *reinterpret_cast<const float4*>(src + i);
  dst[i + 0] = (__bf16)v.x; dst[i + 1] = (__bf16)v.y;
  dst[i + 2] = (__bf16)v.z; dst[i + 3] = (__bf16)v.w;
}

// ---------------------------------------------------------------------------
// Shared GEMM tile staging: 128x32 bf16 tile (8 KB) from row-major src.
// 256 threads, 2 x 16B chunks each, async-to-LDS when available.
// ---------------------------------------------------------------------------
DEVFN void stage_tile(const __bf16* src, int ld, __bf16* lds) {
  const int t = threadIdx.x;
  for (int c = t; c < 512; c += 256) {
    const int row = c >> 2;
    const int col = (c & 3) << 3;
    copy16_g2l(src + (size_t)row * ld + col, lds + row * 32 + col);
  }
}

// ---------------------------------------------------------------------------
// K1: qkv = x @ w_qkv^T  (M=16384, N=2304, K=768), scatter into q/k/v bf16
// [BH][N][64]. WG tile 128x128, 8 waves of 32x64. All-bf16 operands.
// ---------------------------------------------------------------------------
__global__ __launch_bounds__(256)
void nys_gemm_qkv(const __bf16* __restrict__ xb, const __bf16* __restrict__ wq,
                  __bf16* __restrict__ qb, __bf16* __restrict__ kb,
                  __bf16* __restrict__ vb) {
  __shared__ __bf16 As[128 * 32];
  __shared__ __bf16 Bs[128 * 32];
  const int t    = threadIdx.x;
  const int wv   = t >> 5, lane = t & 31;
  const int wm   = wv >> 1, wn = wv & 1;      // wave tile: rows wm*32, cols wn*64
  const int Mb   = blockIdx.y * 128;
  const int Nb   = blockIdx.x * 128;

  v8f acc[2][4];
  for (int i = 0; i < 2; ++i)
    for (int j = 0; j < 4; ++j) acc[i][j] = vzero8();

  const __bf16* xa = xb + (size_t)Mb * C_;
  const __bf16* wa = wq + (size_t)Nb * C_;

  for (int kk = 0; kk < C_; kk += 32) {
    stage_tile(xa + kk, C_, As);
    stage_tile(wa + kk, C_, Bs);
    if (kk + 32 < C_) {
      __builtin_prefetch(xa + (size_t)(t >> 1) * C_ + kk + 32 + (t & 1) * 16, 0, 0);
      __builtin_prefetch(wa + (size_t)(t >> 1) * C_ + kk + 32 + (t & 1) * 16, 0, 0);
    }
    wait_async();
    __syncthreads();
    v16bf af[2], bf[4];
    for (int i = 0; i < 2; ++i) af[i] = load_frag_a(&As[(wm*32 + i*16) * 32], 32);
    for (int j = 0; j < 4; ++j) bf[j] = load_frag_b(&Bs[(wn*64 + j*16) * 32], 32);
    for (int i = 0; i < 2; ++i)
      for (int j = 0; j < 4; ++j) acc[i][j] = wmma_bf16(af[i], bf[j], acc[i][j]);
    __syncthreads();
  }

  const int ln = lane & 15, lh = lane >> 4;
  for (int i = 0; i < 2; ++i) {
    const int m0 = Mb + wm*32 + i*16 + lh*8;   // global token row base
    const int b  = m0 >> 12;                    // token row / 4096
    for (int j = 0; j < 4; ++j) {
      const int cc  = Nb + wn*64 + j*16 + ln;   // 0..2303
      const int sel = cc / C_;
      const int hd  = cc - sel * C_;
      const int h   = hd >> 6, d = hd & 63;
      __bf16* dst = (sel == 0) ? qb : ((sel == 1) ? kb : vb);
      const size_t base = ((size_t)(b * H_ + h) * N_) * D_ + d;
      for (int v = 0; v < 8; ++v) {
        const int n = (m0 + v) & (N_ - 1);
        dst[base + (size_t)n * D_] = (__bf16)acc[i][j][v];
      }
    }
  }
}

// ---------------------------------------------------------------------------
// K2: landmark pooling (mean over 64-token segments) -> f32 + bf16 copies.
// ---------------------------------------------------------------------------
__global__ __launch_bounds__(64)
void nys_pool(const __bf16* __restrict__ qb, const __bf16* __restrict__ kb,
              float* __restrict__ qlm_f, float* __restrict__ klm_f,
              __bf16* __restrict__ qlm_b, __bf16* __restrict__ klm_b) {
  const int bh = blockIdx.x >> 6;
  const int l  = blockIdx.x & 63;
  const int d  = threadIdx.x;
  const __bf16* qp = qb + ((size_t)bh * N_ + l * 64) * D_ + d;
  const __bf16* kp = kb + ((size_t)bh * N_ + l * 64) * D_ + d;
  float sq = 0.f, sk = 0.f;
  for (int i = 0; i < 64; ++i) { sq += (float)qp[(size_t)i * D_]; sk += (float)kp[(size_t)i * D_]; }
  sq *= (1.f / 64.f); sk *= (1.f / 64.f);
  const size_t o = ((size_t)bh * 64 + l) * 64 + d;
  qlm_f[o] = sq; klm_f[o] = sk;
  qlm_b[o] = (__bf16)sq; klm_b[o] = (__bf16)sk;
}

// ---------------------------------------------------------------------------
// K3: inv_mid = inv(softmax(q_lm k_lm^T * scale) + eps*I)  (48 x 64x64, f32)
// One block of 64 threads per matrix; column-parallel Gauss-Jordan in LDS.
// ---------------------------------------------------------------------------
__global__ __launch_bounds__(64)
void nys_midinv(const float* __restrict__ qlm_f, const float* __restrict__ klm_f,
                float* __restrict__ inv_mid) {
  __shared__ float KL[64 * 64];
  __shared__ float Mm[64 * 65];
  __shared__ float Iv[64 * 65];
  __shared__ float colp[64];
  const int bh = blockIdx.x, t = threadIdx.x;

  for (int d = 0; d < 64; ++d) KL[t * 64 + d] = klm_f[((size_t)bh * 64 + t) * 64 + d];
  float qrow[64];
  for (int d = 0; d < 64; ++d) qrow[d] = qlm_f[((size_t)bh * 64 + t) * 64 + d];
  __syncthreads();

  float s[64]; float mx = -3.4e38f;
  for (int c = 0; c < 64; ++c) {
    float dot = 0.f;
    for (int d = 0; d < 64; ++d) dot += qrow[d] * KL[c * 64 + d];
    dot *= SCALE_;
    s[c] = dot; mx = fmaxf(mx, dot);
  }
  float sum = 0.f;
  for (int c = 0; c < 64; ++c) { s[c] = __expf(s[c] - mx); sum += s[c]; }
  const float rinv = 1.f / sum;
  for (int c = 0; c < 64; ++c) {
    Mm[t * 65 + c] = s[c] * rinv + ((c == t) ? EPSREG_ : 0.f);
    Iv[t * 65 + c] = (c == t) ? 1.f : 0.f;
  }
  __syncthreads();

  for (int p = 0; p < 64; ++p) {
    const float invp = 1.f / Mm[p * 65 + p];
    const float mpc  = Mm[p * 65 + t] * invp;
    const float ipc  = Iv[p * 65 + t] * invp;
    __syncthreads();
    Mm[p * 65 + t] = mpc;
    Iv[p * 65 + t] = ipc;
    colp[t] = (t == p) ? 0.f : Mm[t * 65 + p];
    __syncthreads();
    for (int r2 = 0; r2 < 64; ++r2) {
      const float f = colp[r2];
      Mm[r2 * 65 + t] -= f * mpc;
      Iv[r2 * 65 + t] -= f * ipc;
    }
    __syncthreads();
  }
  for (int c = 0; c < 64; ++c)
    inv_mid[((size_t)bh * 64 + t) * 64 + c] = Iv[t * 65 + c];
}

// ---------------------------------------------------------------------------
// K4a: S[bh][l][n] = scale * (q_lm @ k^T)   (WMMA). grid (N/128, BH).
// ---------------------------------------------------------------------------
__global__ __launch_bounds__(256)
void nys_s_qmk(const __bf16* __restrict__ qlm_b, const __bf16* __restrict__ kb,
               float* __restrict__ S) {
  const int bh = blockIdx.y;
  const int n0 = blockIdx.x * 128;
  const int wv = threadIdx.x >> 5;
  const int mb = wv & 3, nh = wv >> 2;
  const __bf16* qa  = qlm_b + (size_t)bh * 64 * 64 + mb * 16 * 64;
  const __bf16* kbp = kb + ((size_t)bh * N_ + n0 + nh * 64) * 64;

  v8f acc[4];
  for (int j = 0; j < 4; ++j) acc[j] = vzero8();
  for (int ks = 0; ks < 2; ++ks) {
    v16bf a = load_frag_a(qa + ks * 32, 64);
    for (int j = 0; j < 4; ++j)
      acc[j] = wmma_bf16(a, load_frag_b(kbp + j * 16 * 64 + ks * 32, 64), acc[j]);
  }
  const int lane = threadIdx.x & 31, ln = lane & 15, lh = lane >> 4;
  for (int j = 0; j < 4; ++j) {
    float* dst = S + ((size_t)bh * 64 + mb * 16 + lh * 8) * N_ + n0 + nh * 64 + j * 16 + ln;
    for (int v = 0; v < 8; ++v) dst[(size_t)v * N_] = acc[j][v] * SCALE_;
  }
}

// ---------------------------------------------------------------------------
// K4b: per-row max & exp-sum over N=4096. grid (64, BH), block 128.
// ---------------------------------------------------------------------------
__global__ __launch_bounds__(128)
void nys_rowstat(const float* __restrict__ S, float* __restrict__ rowmax,
                 float* __restrict__ rowsum) {
  __shared__ float red[128];
  const int l = blockIdx.x, bh = blockIdx.y, t = threadIdx.x;
  const float* row = S + ((size_t)bh * 64 + l) * N_;
  float m = -3.4e38f;
  for (int i = t; i < N_; i += 128) m = fmaxf(m, row[i]);
  red[t] = m; __syncthreads();
  for (int s = 64; s > 0; s >>= 1) { if (t < s) red[t] = fmaxf(red[t], red[t + s]); __syncthreads(); }
  m = red[0]; __syncthreads();
  float sum = 0.f;
  for (int i = t; i < N_; i += 128) sum += __expf(row[i] - m);
  red[t] = sum; __syncthreads();
  for (int s = 64; s > 0; s >>= 1) { if (t < s) red[t] += red[t + s]; __syncthreads(); }
  if (t == 0) { rowmax[bh * 64 + l] = m; rowsum[bh * 64 + l] = red[0]; }
}

// ---------------------------------------------------------------------------
// K4c: P = bf16(exp(S - rowmax)/rowsum). Elementwise, 4 per thread.
// ---------------------------------------------------------------------------
__global__ __launch_bounds__(256)
void nys_expnorm(const float* __restrict__ S, const float* __restrict__ rowmax,
                 const float* __restrict__ rowsum, __bf16* __restrict__ P) {
  const size_t base = ((size_t)blockIdx.x * 256 + threadIdx.x) * 4;
  const int row = (int)(base >> 12);
  const float m = rowmax[row], rinv = 1.f / rowsum[row];
  const float4 s4 = *reinterpret_cast<const float4*>(S + base);
  P[base + 0] = (__bf16)(__expf(s4.x - m) * rinv);
  P[base + 1] = (__bf16)(__expf(s4.y - m) * rinv);
  P[base + 2] = (__bf16)(__expf(s4.z - m) * rinv);
  P[base + 3] = (__bf16)(__expf(s4.w - m) * rinv);
}

// ---------------------------------------------------------------------------
// K4d: T1 = P @ v  (M=64, N=64, K=4096) per bh. v transposed through LDS.
// grid = BH, 8 waves: wave = (mb rows, dh 32-col half).
// ---------------------------------------------------------------------------
__global__ __launch_bounds__(256)
void nys_gemm_pv(const __bf16* __restrict__ P, const __bf16* __restrict__ vb,
                 float* __restrict__ T1) {
  __shared__ __bf16 vt[64 * 32];   // [d][n_local]
  const int bh = blockIdx.x;
  const int t  = threadIdx.x, wv = t >> 5;
  const int mb = wv & 3, dh = wv >> 2;
  v8f acc[2]; acc[0] = vzero8(); acc[1] = vzero8();

  const int nl = t >> 3;           // 0..31 (row of v tile)
  const int dc = (t & 7) * 8;      // d chunk

  for (int kk = 0; kk < N_; kk += 32) {
    uint4 raw = *reinterpret_cast<const uint4*>(vb + ((size_t)bh * N_ + kk + nl) * 64 + dc);
    __bf16 tmp[8];
    *reinterpret_cast<uint4*>(tmp) = raw;
    for (int j = 0; j < 8; ++j) vt[(dc + j) * 32 + nl] = tmp[j];
    __syncthreads();
    v16bf a = load_frag_a(P + ((size_t)bh * 64 + mb * 16) * N_ + kk, N_);
    for (int j = 0; j < 2; ++j)
      acc[j] = wmma_bf16(a, load_frag_b(&vt[(dh * 32 + j * 16) * 32], 32), acc[j]);
    __syncthreads();
  }
  for (int j = 0; j < 2; ++j)
    store_cfrag_f32(T1 + ((size_t)bh * 64 + mb * 16) * 64 + dh * 32 + j * 16, 64, acc[j]);
}

// ---------------------------------------------------------------------------
// K4e: T2t[bh][d][l] = (inv_mid @ T1)[l][d] as bf16 (transposed for B-operand).
// ---------------------------------------------------------------------------
__global__ __launch_bounds__(256)
void nys_t2(const float* __restrict__ inv_mid, const float* __restrict__ T1,
            __bf16* __restrict__ T2t) {
  const int bh = blockIdx.x, t = threadIdx.x;
  for (int e = t; e < 4096; e += 256) {
    const int d = e >> 6, l = e & 63;
    float acc = 0.f;
    for (int m = 0; m < 64; ++m)
      acc += inv_mid[((size_t)bh * 64 + l) * 64 + m] * T1[((size_t)bh * 64 + m) * 64 + d];
    T2t[(size_t)bh * 4096 + d * 64 + l] = (__bf16)acc;
  }
}

// ---------------------------------------------------------------------------
// K5: out_pre[b][n][h*64+d] = softmax(q @ k_lm^T * scale) @ T2  (fused, WMMA).
// grid (N/128, BH); 8 waves x 16 token rows. Row softmax via shfl_xor(...,16).
// ---------------------------------------------------------------------------
__global__ __launch_bounds__(256)
void nys_attn_out(const __bf16* __restrict__ qb, const __bf16* __restrict__ klm_b,
                  const __bf16* __restrict__ t2t, __bf16* __restrict__ out_pre) {
  __shared__ __bf16 Pl[8 * 16 * 80];   // per-wave 16x64 P tile, ld=80 (160B rows)
  const int bh = blockIdx.y;
  const int n0 = blockIdx.x * 128;
  const int t  = threadIdx.x, wv = t >> 5, lane = t & 31;
  const int ln = lane & 15, lh = lane >> 4;
  const int row0 = n0 + wv * 16;

  const __bf16* qa = qb + ((size_t)bh * N_ + row0) * 64;
  const __bf16* kl = klm_b + (size_t)bh * 64 * 64;
  v8f s[4];
  for (int j = 0; j < 4; ++j) s[j] = vzero8();
  for (int ks = 0; ks < 2; ++ks) {
    v16bf a = load_frag_a(qa + ks * 32, 64);
    for (int j = 0; j < 4; ++j)
      s[j] = wmma_bf16(a, load_frag_b(kl + j * 16 * 64 + ks * 32, 64), s[j]);
  }

  __bf16* pw = &Pl[wv * 16 * 80];
  for (int v = 0; v < 8; ++v) {
    float m = -3.4e38f;
    for (int j = 0; j < 4; ++j) { s[j][v] *= SCALE_; m = fmaxf(m, s[j][v]); }
    for (int off = 1; off < 16; off <<= 1) m = fmaxf(m, __shfl_xor(m, off, 16));
    float e[4]; float sum = 0.f;
    for (int j = 0; j < 4; ++j) { e[j] = __expf(s[j][v] - m); sum += e[j]; }
    for (int off = 1; off < 16; off <<= 1) sum += __shfl_xor(sum, off, 16);
    const float rinv = 1.f / sum;
    const int prow = v + lh * 8;
    for (int j = 0; j < 4; ++j) pw[prow * 80 + j * 16 + ln] = (__bf16)(e[j] * rinv);
  }
  __syncthreads();

  v8f o[4];
  for (int j = 0; j < 4; ++j) o[j] = vzero8();
  const __bf16* t2 = t2t + (size_t)bh * 4096;   // [d][l]
  for (int ks = 0; ks < 2; ++ks) {
    v16bf a = load_frag_a(pw + ks * 32, 80);
    for (int j = 0; j < 4; ++j)
      o[j] = wmma_bf16(a, load_frag_b(t2 + j * 16 * 64 + ks * 32, 64), o[j]);
  }
  const int b = bh / H_, h = bh % H_;
  for (int j = 0; j < 4; ++j) {
    const int c = h * 64 + j * 16 + ln;
    const size_t base = ((size_t)b * N_ + row0 + lh * 8) * C_ + c;
    for (int v = 0; v < 8; ++v)
      out_pre[base + (size_t)v * C_] = (__bf16)o[j][v];
  }
}

// ---------------------------------------------------------------------------
// K6: out = out_pre @ w_proj^T + b_proj  (M=16384, N=768, K=768), f32 output.
// Both operands bf16, staged via async-to-LDS.
// ---------------------------------------------------------------------------
__global__ __launch_bounds__(256)
void nys_gemm_proj(const __bf16* __restrict__ A, const __bf16* __restrict__ wp,
                   const float* __restrict__ bias, float* __restrict__ out) {
  __shared__ __bf16 As[128 * 32];
  __shared__ __bf16 Bs[128 * 32];
  const int t  = threadIdx.x;
  const int wv = t >> 5, lane = t & 31;
  const int wm = wv >> 1, wn = wv & 1;
  const int Mb = blockIdx.y * 128;
  const int Nb = blockIdx.x * 128;

  v8f acc[2][4];
  for (int i = 0; i < 2; ++i)
    for (int j = 0; j < 4; ++j) acc[i][j] = vzero8();

  const __bf16* aa = A + (size_t)Mb * C_;
  const __bf16* wa = wp + (size_t)Nb * C_;

  for (int kk = 0; kk < C_; kk += 32) {
    stage_tile(aa + kk, C_, As);
    stage_tile(wa + kk, C_, Bs);
    if (kk + 32 < C_) {
      __builtin_prefetch(aa + (size_t)(t >> 1) * C_ + kk + 32 + (t & 1) * 16, 0, 0);
      __builtin_prefetch(wa + (size_t)(t >> 1) * C_ + kk + 32 + (t & 1) * 16, 0, 0);
    }
    wait_async();
    __syncthreads();
    v16bf af[2], bf[4];
    for (int i = 0; i < 2; ++i) af[i] = load_frag_a(&As[(wm*32 + i*16) * 32], 32);
    for (int j = 0; j < 4; ++j) bf[j] = load_frag_b(&Bs[(wn*64 + j*16) * 32], 32);
    for (int i = 0; i < 2; ++i)
      for (int j = 0; j < 4; ++j) acc[i][j] = wmma_bf16(af[i], bf[j], acc[i][j]);
    __syncthreads();
  }

  const int ln = lane & 15, lh = lane >> 4;
  for (int i = 0; i < 2; ++i) {
    const int m0 = Mb + wm*32 + i*16 + lh*8;
    for (int j = 0; j < 4; ++j) {
      const int cc = Nb + wn*64 + j*16 + ln;
      const float bv = bias[cc];
      for (int v = 0; v < 8; ++v)
        out[(size_t)(m0 + v) * C_ + cc] = acc[i][j][v] + bv;
    }
  }
}

// ---------------------------------------------------------------------------
// Launch
// ---------------------------------------------------------------------------
extern "C" void kernel_launch(void* const* d_in, const int* in_sizes, int n_in,
                              void* d_out, int out_size, void* d_ws, size_t ws_size,
                              hipStream_t stream) {
  (void)in_sizes; (void)n_in; (void)out_size; (void)ws_size;
  const float* x      = (const float*)d_in[0];
  const float* w_qkv  = (const float*)d_in[1];
  const float* w_proj = (const float*)d_in[2];
  const float* b_proj = (const float*)d_in[3];
  float* out = (float*)d_out;

  char* ws = (char*)d_ws;
  size_t off = 0;
  auto alloc = [&](size_t bytes) -> void* {
    void* p = ws + off;
    off = (off + bytes + 255) & ~(size_t)255;
    return p;
  };
  const size_t bhnd = (size_t)BH_ * N_ * D_;        // 12.58M elems
  __bf16* qb      = (__bf16*)alloc(bhnd * 2);
  __bf16* kb      = (__bf16*)alloc(bhnd * 2);
  __bf16* vb      = (__bf16*)alloc(bhnd * 2);
  __bf16* out_pre = (__bf16*)alloc((size_t)B_ * N_ * C_ * 2);
  float*  S       = (float*) alloc((size_t)BH_ * 64 * N_ * 4);
  __bf16* P       = (__bf16*)alloc((size_t)BH_ * 64 * N_ * 2);
  __bf16* xbf     = (__bf16*)alloc((size_t)B_ * N_ * C_ * 2);
  __bf16* wqkv_bf = (__bf16*)alloc((size_t)3 * C_ * C_ * 2);
  __bf16* wproj_bf= (__bf16*)alloc((size_t)C_ * C_ * 2);
  float*  qlm_f   = (float*) alloc((size_t)BH_ * 64 * 64 * 4);
  float*  klm_f   = (float*) alloc((size_t)BH_ * 64 * 64 * 4);
  __bf16* qlm_b   = (__bf16*)alloc((size_t)BH_ * 64 * 64 * 2);
  __bf16* klm_b   = (__bf16*)alloc((size_t)BH_ * 64 * 64 * 2);
  float*  inv_mid = (float*) alloc((size_t)BH_ * 64 * 64 * 4);
  float*  T1      = (float*) alloc((size_t)BH_ * 64 * 64 * 4);
  __bf16* T2t     = (__bf16*)alloc((size_t)BH_ * 64 * 64 * 2);
  float*  rowmax  = (float*) alloc((size_t)BH_ * 64 * 4);
  float*  rowsum  = (float*) alloc((size_t)BH_ * 64 * 4);

  // f32 -> bf16 one-shot conversions
  nys_cvt_bf16 <<<dim3((B_ * N_ * C_) / 1024), 256, 0, stream>>>(x, xbf);
  nys_cvt_bf16 <<<dim3((3 * C_ * C_) / 1024), 256, 0, stream>>>(w_qkv, wqkv_bf);
  nys_cvt_bf16 <<<dim3((C_ * C_) / 1024), 256, 0, stream>>>(w_proj, wproj_bf);

  nys_gemm_qkv <<<dim3(18, 128), 256, 0, stream>>>(xbf, wqkv_bf, qb, kb, vb);
  nys_pool     <<<dim3(BH_ * 64), 64, 0, stream>>>(qb, kb, qlm_f, klm_f, qlm_b, klm_b);
  nys_midinv   <<<dim3(BH_), 64, 0, stream>>>(qlm_f, klm_f, inv_mid);
  nys_s_qmk    <<<dim3(N_ / 128, BH_), 256, 0, stream>>>(qlm_b, kb, S);
  nys_rowstat  <<<dim3(64, BH_), 128, 0, stream>>>(S, rowmax, rowsum);
  nys_expnorm  <<<dim3((BH_ * 64 * N_) / 1024), 256, 0, stream>>>(S, rowmax, rowsum, P);
  nys_gemm_pv  <<<dim3(BH_), 256, 0, stream>>>(P, vb, T1);
  nys_t2       <<<dim3(BH_), 256, 0, stream>>>(inv_mid, T1, T2t);
  nys_attn_out <<<dim3(N_ / 128, BH_), 256, 0, stream>>>(qb, klm_b, T2t, out_pre);
  nys_gemm_proj<<<dim3(C_ / 128, 16384 / 128), 256, 0, stream>>>(out_pre, wproj_bf, b_proj, out);
}
